// OSAttention_14181982011827
// MI455X (gfx1250) — compile-verified, matrix operand order checked by reading, not run on hardware
//
#include <hip/hip_runtime.h>
#include <hip/hip_bf16.h>

typedef _Float16 v16h __attribute__((ext_vector_type(16)));
typedef _Float16 v8h  __attribute__((ext_vector_type(8)));
typedef _Float16 v4h  __attribute__((ext_vector_type(4)));
typedef float    v8f  __attribute__((ext_vector_type(8)));
typedef float    v4f  __attribute__((ext_vector_type(4)));
typedef unsigned int v4u __attribute__((ext_vector_type(4)));
typedef int      v4i  __attribute__((ext_vector_type(4)));
typedef int      v8i  __attribute__((ext_vector_type(8)));

#define DMODEL 512
#define NHEAD  8
#define DKH    64     // per-head dim
#define NKEY   64     // keys per query
#define BQ     4096   // B * NQ

#if defined(__has_builtin)
#  if __has_builtin(__builtin_amdgcn_global_load_async_to_lds_b128)
#    define HAVE_ASYNC_LDS 1
#  endif
#  if __has_builtin(__builtin_amdgcn_s_wait_asynccnt)
#    define HAVE_ASYNCCNT 1
#  endif
#endif

__device__ __forceinline__ v8f wmma16x16x32(v16h a, v16h b, v8f c) {
  // D = A(16x32 f16) x B(32x16 f16) + C(16x16 f32)
  return __builtin_amdgcn_wmma_f32_16x16x32_f16(false, a, false, b, (short)0, c,
                                                false, false);
}

// --- CDNA5 async global->LDS copy of one 16B chunk (ASYNCcnt tracked) ------
// Toolchain signature (from diagnostics): params are pointers to int4.
__device__ __forceinline__ void async_copy16(const _Float16* g, _Float16* l) {
#if HAVE_ASYNC_LDS
  __builtin_amdgcn_global_load_async_to_lds_b128((v4i*)(void*)g, (v4i*)l, 0, 0);
#else
  *(v8h*)l = *(const v8h*)g;
#endif
}
__device__ __forceinline__ void wait_async_all() {
#if HAVE_ASYNC_LDS
#  if HAVE_ASYNCCNT
  __builtin_amdgcn_s_wait_asynccnt(0);
#  else
  asm volatile("s_wait_asynccnt 0x0" ::: "memory");
#  endif
#endif
}

// --- CDNA5 Tensor Data Mover: DMA a 2D f16 tile (tile_w x tile_h, row stride
// tensor_w elements) from global into LDS at byte offset lds_off, padding the
// LDS destination by 4 DWORDs after every 16 DWORDs (i.e. 32-half rows land
// on a 40-half pitch -> conflict-free b128 fragment reads). One wave issues
// one descriptor; completion via TENSORcnt.  D# packing per ISA 08 §8.3/8.4.
__device__ __forceinline__ void tdm_load_tile_f16(const _Float16* gsrc,
                                                  unsigned lds_off,
                                                  unsigned tensor_w,
                                                  unsigned tile_w,
                                                  unsigned tile_h) {
  const unsigned long long ga = (unsigned long long)(size_t)gsrc;
  v4u g0;
  g0[0] = 1u;                                   // count=1, user descriptor
  g0[1] = lds_off;                              // lds_addr
  g0[2] = (unsigned)(ga & 0xFFFFFFFFu);         // global_addr[31:0]
  g0[3] = (unsigned)((ga >> 32) & 0x01FFFFFFu)  // global_addr[56:32]
          | (2u << 30);                         // type = 2 ("image")
  v8i g1;
  // data_size=2B (code 1), pad_enable=1, pad_interval code 3 (16 DWORDs),
  // pad_amount code 3 (4 DWORDs); workgroup_mask=0 (not in a cluster)
  g1[0] = (int)((1u << 16) | (1u << 20) | (3u << 22) | (3u << 25));
  g1[1] = (int)((tensor_w & 0xFFFFu) << 16);    // tensor_dim0[15:0]
  g1[2] = (int)(((tensor_w >> 16) & 0xFFFFu) |  // tensor_dim0[31:16]
                ((tensor_w & 0xFFFFu) << 16));  // tensor_dim1[15:0] (=tensor_w)
  g1[3] = (int)(((tensor_w >> 16) & 0xFFFFu) |  // tensor_dim1[31:16]
                (tile_w << 16));                // tile_dim0
  g1[4] = (int)(tile_h & 0xFFFFu);              // tile_dim1 (tile_dim2 = 0)
  g1[5] = (int)tensor_w;                        // tensor_dim0_stride[31:0]
  g1[6] = 0;                                    // stride0 hi16 / stride1 lo16
  g1[7] = 0;                                    // stride1 hi32
  asm volatile("tensor_load_to_lds %0, %1" ::"s"(g0), "s"(g1) : "memory");
}
__device__ __forceinline__ void wait_tensor_all() {
  __builtin_amdgcn_s_wait_tensorcnt(0);
}

// Build a 16x32 f16 fragment (A or B operand) from an LDS tile with padded
// row stride of 40 halves.  Lanes 0-15: k={0..7,16..23}; lanes 16-31:
// k={8..15,24..31} of row (tile16*16+lo), per the 16-bit WMMA VGPR layout.
__device__ __forceinline__ v16h frag_lds40(const _Float16* tile, int tile16,
                                           int lo, int hi) {
  const _Float16* p = tile + (tile16 * 16 + lo) * 40 + hi * 8;
  v8h a0 = *(const v8h*)p;
  v8h a1 = *(const v8h*)(p + 16);
  v16h f;
#pragma unroll
  for (int i = 0; i < 8; ++i) { f[i] = a0[i]; f[i + 8] = a1[i]; }
  return f;
}

// Build a 16x32 A fragment straight from a row-major f32 matrix (k-contig),
// converting f32 -> f16 in-register. colbase = kstep*32 + hi*8.
__device__ __forceinline__ v16h frag_gf32(const float* __restrict__ X, int row,
                                          int colbase) {
  const float* p = X + (size_t)row * DMODEL + colbase;
  v16h f;
#pragma unroll
  for (int i = 0; i < 8; ++i) f[i] = (_Float16)p[i];
#pragma unroll
  for (int i = 0; i < 8; ++i) f[i + 8] = (_Float16)p[16 + i];
  return f;
}

// ---------------------------------------------------------------------------
// Weight conversion: f32 [n][k] -> f16 [n][k] (already B-operand friendly).
// ---------------------------------------------------------------------------
__global__ __launch_bounds__(256) void cvt_f32_to_f16(
    _Float16* __restrict__ dst, const float* __restrict__ src, int n) {
  int i = (blockIdx.x * 256 + threadIdx.x) * 4;
  if (i + 3 < n) {
    v4f x = *(const v4f*)(src + i);
    v4h h;
    h[0] = (_Float16)x[0]; h[1] = (_Float16)x[1];
    h[2] = (_Float16)x[2]; h[3] = (_Float16)x[3];
    *(v4h*)(dst + i) = h;
  }
}

// ---------------------------------------------------------------------------
// Generic C[M][512] = A[M][512] @ Bw^T + bias.  Bw is f16 [512 n][512 k].
// Block tile 128x128, BK=32, 8 waves: wave (mw,nw) = (w>>2, w&3) computes a
// 64x32 sub-tile = 4x2 WMMA accumulators.  A staged per-thread (f32 converts
// in-register, f16 uses async-to-LDS); B tiles DMA'd by the TDM, both
// double-buffered against the WMMA inner loop.
// ---------------------------------------------------------------------------
template <typename AT, bool OUT_F32>
__global__ __launch_bounds__(256) void gemm512_wmma(
    const AT* __restrict__ A, const _Float16* __restrict__ Bw,
    const float* __restrict__ bias, void* __restrict__ Cout, int M) {
  __shared__ alignas(16) _Float16 As[2][128][40];
  __shared__ alignas(16) _Float16 Bs[2][128][40];
  const int m0 = blockIdx.x * 128;
  const int n0 = blockIdx.y * 128;
  const int t = threadIdx.x;
  const int lane = t & 31, w = t >> 5;
  const int lo = lane & 15, hi = lane >> 4;
  const int mw = w >> 2, nw = w & 3;

  auto stage_A = [&](int kt, int buf) {
    const int r = t >> 1, hf = t & 1;
    const int gk = kt * 32 + hf * 16;
    if constexpr (sizeof(AT) == 4) {  // f32 source: convert while staging
      const float* src = (const float*)A + (size_t)(m0 + r) * DMODEL + gk;
#pragma unroll
      for (int i = 0; i < 4; ++i) {
        v4f x = *(const v4f*)(src + i * 4);
        v4h h;
        h[0] = (_Float16)x[0]; h[1] = (_Float16)x[1];
        h[2] = (_Float16)x[2]; h[3] = (_Float16)x[3];
        *(v4h*)&As[buf][r][hf * 16 + i * 4] = h;
      }
    } else {  // f16 source: async global->LDS, no VGPR round-trip
      const _Float16* src = (const _Float16*)A + (size_t)(m0 + r) * DMODEL + gk;
      async_copy16(src, &As[buf][r][hf * 16]);
      async_copy16(src + 8, &As[buf][r][hf * 16 + 8]);
    }
  };
  auto issue_B = [&](int kt, int buf) {  // one TDM descriptor per tile
    if (w == 0)
      tdm_load_tile_f16(Bw + (size_t)n0 * DMODEL + kt * 32,
                        (unsigned)(size_t)&Bs[buf][0][0], DMODEL, 32, 128);
  };
  auto publish = [&]() {  // make staged tiles visible to all waves
    wait_async_all();
    if (w == 0) wait_tensor_all();
    __syncthreads();
  };

  v8f acc[4][2] = {};
  stage_A(0, 0);
  issue_B(0, 0);
  publish();
  for (int kt = 0; kt < 16; ++kt) {
    const int cur = kt & 1;
    if (kt + 1 < 16) {
      stage_A(kt + 1, cur ^ 1);
      issue_B(kt + 1, cur ^ 1);
    }
    v16h af[4], bf[2];
#pragma unroll
    for (int i = 0; i < 4; ++i)
      af[i] = frag_lds40(&As[cur][0][0], 4 * mw + i, lo, hi);
#pragma unroll
    for (int j = 0; j < 2; ++j)
      bf[j] = frag_lds40(&Bs[cur][0][0], 2 * nw + j, lo, hi);
#pragma unroll
    for (int i = 0; i < 4; ++i)
#pragma unroll
      for (int j = 0; j < 2; ++j)
        acc[i][j] = wmma16x16x32(af[i], bf[j], acc[i][j]);
    publish();
  }
  // Epilogue. C/D layout: VGPR r holds row r (lanes 0-15) / row 8+r (16-31),
  // column = lane&15 within the 16x16 tile.
#pragma unroll
  for (int i = 0; i < 4; ++i) {
    const int mbase = m0 + (4 * mw + i) * 16 + hi * 8;
#pragma unroll
    for (int j = 0; j < 2; ++j) {
      const int ncol = n0 + (2 * nw + j) * 16 + lo;
      const float bb = bias[ncol];
#pragma unroll
      for (int r = 0; r < 8; ++r) {
        const float val = acc[i][j][r] + bb;
        if constexpr (OUT_F32)
          ((float*)Cout)[(size_t)(mbase + r) * DMODEL + ncol] = val;
        else
          ((_Float16*)Cout)[(size_t)(mbase + r) * DMODEL + ncol] = (_Float16)val;
      }
    }
  }
}

// ---------------------------------------------------------------------------
// Fused per-(b,q) kernel: project K and V tiles (64x512 @ 512x512 WMMA) into
// LDS, then scores -> softmax -> P@V. 4096 blocks, 256 threads (8 waves).
// Keys/values f32 are read from HBM exactly once; weight tiles are TDM-DMA'd
// and stay L2-resident across all 4096 blocks.
// ---------------------------------------------------------------------------
__global__ __launch_bounds__(256) void fused_kv_attention(
    const float* __restrict__ keys, const float* __restrict__ values,
    const _Float16* __restrict__ wk16, const _Float16* __restrict__ wv16,
    const float* __restrict__ bk, const float* __restrict__ bv,
    const _Float16* __restrict__ qp, _Float16* __restrict__ attout) {
  __shared__ alignas(16) _Float16 P[2][64][520];   // projected K, V (padded)
  __shared__ alignas(16) _Float16 Bs[2][128][40];  // weight tile, dbl-buffered
  __shared__ alignas(16) float sc[NHEAD][NKEY];    // scores -> probs
  __shared__ alignas(16) _Float16 qrow[DMODEL];    // projected q, all heads

  const int g = blockIdx.x;  // b*NQ + q
  const int t = threadIdx.x;
  const int lane = t & 31, w = t >> 5;
  const int lo = lane & 15, hi = lane >> 4;
  const int mt = w & 3;         // m-tile (16 key-rows) handled by this wave
  const int nb = (w >> 2) * 4;  // first of 4 n-tiles handled by this wave

  if (t < 64)
    async_copy16(qp + (size_t)g * DMODEL + t * 8, &qrow[t * 8]);
  wait_async_all();

  // Warm L2 with the values tile while K is being projected.
  __builtin_prefetch(values + (size_t)g * (NKEY * DMODEL) + t * 128, 0, 1);

  for (int s = 0; s < 2; ++s) {
    const float* X = (s == 0 ? keys : values) + (size_t)g * (NKEY * DMODEL);
    const _Float16* W = (s == 0) ? wk16 : wv16;
    const float* bias = (s == 0) ? bk : bv;

    for (int nblk = 0; nblk < 4; ++nblk) {
      const int n0 = nblk * 128;
      auto issue_B = [&](int kt, int buf) {
        if (w == 0)
          tdm_load_tile_f16(W + (size_t)n0 * DMODEL + kt * 32,
                            (unsigned)(size_t)&Bs[buf][0][0], DMODEL, 32, 128);
      };
      auto publish = [&]() {
        if (w == 0) wait_tensor_all();
        __syncthreads();
      };
      v8f acc[4] = {};
      issue_B(0, 0);
      publish();
      for (int kt = 0; kt < 16; ++kt) {
        const int cur = kt & 1;
        if (kt + 1 < 16) issue_B(kt + 1, cur ^ 1);
        v16h af = frag_gf32(X, mt * 16 + lo, kt * 32 + hi * 8);
#pragma unroll
        for (int j = 0; j < 4; ++j) {
          v16h bf = frag_lds40(&Bs[cur][0][0], nb + j, lo, hi);
          acc[j] = wmma16x16x32(af, bf, acc[j]);
        }
        publish();
      }
#pragma unroll
      for (int j = 0; j < 4; ++j) {
        const int ncol = n0 + (nb + j) * 16 + lo;
        const float bb = bias[ncol];
#pragma unroll
        for (int r = 0; r < 8; ++r)
          P[s][mt * 16 + hi * 8 + r][ncol] = (_Float16)(acc[j][r] + bb);
      }
      __syncthreads();
    }
  }
  __syncthreads();

  // scores[h][k] = (q_h . k_kh) / sqrt(64)
  for (int p = t; p < NHEAD * NKEY; p += 256) {
    const int h = p >> 6, k = p & 63;
    float s1 = 0.f;
    const _Float16* qh = &qrow[h * DKH];
    const _Float16* kr = &P[0][k][h * DKH];
#pragma unroll 8
    for (int d = 0; d < DKH; ++d) s1 += (float)qh[d] * (float)kr[d];
    sc[h][k] = s1 * 0.125f;
  }
  __syncthreads();
  {  // softmax over 64 keys: wave w owns head w (wave32 shuffle reductions)
    float v0 = sc[w][lane], v1 = sc[w][lane + 32];
    float m = fmaxf(v0, v1);
#pragma unroll
    for (int off = 16; off > 0; off >>= 1)
      m = fmaxf(m, __shfl_xor(m, off, 32));
    const float e0 = __expf(v0 - m), e1 = __expf(v1 - m);
    float sum = e0 + e1;
#pragma unroll
    for (int off = 16; off > 0; off >>= 1) sum += __shfl_xor(sum, off, 32);
    const float inv = 1.f / sum;
    sc[w][lane] = e0 * inv;
    sc[w][lane + 32] = e1 * inv;
  }
  __syncthreads();
  // out[h][d] = sum_k p[h][k] * v[k][h*64+d]
  for (int p = t; p < NHEAD * DKH; p += 256) {
    const int h = p >> 6, d = p & 63;
    float o = 0.f;
#pragma unroll 8
    for (int k = 0; k < NKEY; ++k) o += sc[h][k] * (float)P[1][k][h * DKH + d];
    attout[(size_t)g * DMODEL + h * DKH + d] = (_Float16)o;
  }
}

// ---------------------------------------------------------------------------
extern "C" void kernel_launch(void* const* d_in, const int* in_sizes, int n_in,
                              void* d_out, int out_size, void* d_ws,
                              size_t ws_size, hipStream_t stream) {
  const float* queries = (const float*)d_in[0];
  const float* keys    = (const float*)d_in[1];
  const float* values  = (const float*)d_in[2];
  const float* Wq = (const float*)d_in[3];  const float* bq = (const float*)d_in[4];
  const float* Wk = (const float*)d_in[5];  const float* bk = (const float*)d_in[6];
  const float* Wv = (const float*)d_in[7];  const float* bv = (const float*)d_in[8];
  const float* Wo = (const float*)d_in[9];  const float* bo = (const float*)d_in[10];
  float* out = (float*)d_out;

  // Workspace layout (all f16): 4 weight matrices (512KB each) + qp + attout
  // = ~10 MB total.
  _Float16* wq16 = (_Float16*)d_ws;
  _Float16* wk16 = wq16 + 512 * 512;
  _Float16* wv16 = wk16 + 512 * 512;
  _Float16* wo16 = wv16 + 512 * 512;
  _Float16* qp     = wo16 + 512 * 512;           // [4096][512]
  _Float16* attout = qp + (size_t)BQ * DMODEL;   // [4096][512]

  cvt_f32_to_f16<<<256, 256, 0, stream>>>(wq16, Wq, 512 * 512);
  cvt_f32_to_f16<<<256, 256, 0, stream>>>(wk16, Wk, 512 * 512);
  cvt_f32_to_f16<<<256, 256, 0, stream>>>(wv16, Wv, 512 * 512);
  cvt_f32_to_f16<<<256, 256, 0, stream>>>(wo16, Wo, 512 * 512);

  // Q projection: [4096][512] = queries @ Wq^T + bq  (f16 out)
  gemm512_wmma<float, false>
      <<<dim3(32, 4), 256, 0, stream>>>(queries, wq16, bq, (void*)qp, BQ);

  // Fused K/V projection + local attention, one block per (b,q)
  fused_kv_attention<<<BQ, 256, 0, stream>>>(keys, values, wk16, wv16, bk, bv,
                                             qp, attout);

  // Output projection: out = attout @ Wo^T + bo  (f32 out)
  gemm512_wmma<_Float16, true>
      <<<dim3(32, 4), 256, 0, stream>>>(attout, wo16, bo, (void*)out, BQ);
}
